// Processor_1262720385182
// MI455X (gfx1250) — matrix operand order repaired and checked by reading, not compiled, not run on hardware
//
#include <hip/hip_runtime.h>

#define NB 2
#define NN 20000
#define NE 160000
#define DD 128
#define NS 6

typedef unsigned int u32;
typedef unsigned short u16;
typedef __attribute__((ext_vector_type(16))) __bf16 bf16x16;
typedef __attribute__((ext_vector_type(8)))  float  f32x8;
typedef __attribute__((ext_vector_type(4)))  float  f32x4;
typedef __attribute__((ext_vector_type(4)))  u32    u32x4;

union Frag16 { bf16x16 v; u32x4 q[2]; };

__device__ __forceinline__ f32x8 wmma_bf16(bf16x16 a, bf16x16 b, f32x8 c) {
  return __builtin_amdgcn_wmma_f32_16x16x32_bf16(false, a, false, b, (short)0, c,
                                                 false, false);
}

__device__ __forceinline__ u32 pkbf(float lo, float hi) {
  u16 a = __builtin_bit_cast(u16, (__bf16)lo);
  u16 b = __builtin_bit_cast(u16, (__bf16)hi);
  return (u32)a | ((u32)b << 16);
}

// Fill 8 consecutive-K elements of an A fragment from 8 contiguous f32 (scaled).
__device__ __forceinline__ void setA8(bf16x16& a, int base, const float* p, float sc) {
  f32x4 x = *(const f32x4*)p;
  f32x4 y = *(const f32x4*)(p + 4);
#pragma unroll
  for (int j = 0; j < 4; ++j) {
    a[base + j]     = (__bf16)(x[j] * sc);
    a[base + 4 + j] = (__bf16)(y[j] * sc);
  }
}

// swish via hardware v_exp_f32 + v_rcp_f32 (avoid IEEE div expansion)
__device__ __forceinline__ float swishf(float v) {
  return v * __builtin_amdgcn_rcpf(1.0f + __expf(-v));
}

// ---------------------------------------------------------------------------
// Pack an f32 weight [K][128] into per-lane WMMA B-fragment dwords (bf16 pairs)
// dst[((kt*8+nt)*32+lane)*8 + i] = pack(W[(32kt+kb+2i+(i>=4?8:0))*128+n], W[(..+1)*128+n])
// kb = (lane>=16)?8:0, n = (lane&15)+16*nt
// ---------------------------------------------------------------------------
__global__ void pack_w_kernel(const float* __restrict__ src, u32* __restrict__ dst, int K) {
  int idx = blockIdx.x * blockDim.x + threadIdx.x;
  int total = K * 64;
  if (idx >= total) return;
  int i    = idx & 7;
  int lane = (idx >> 3) & 31;
  int nt   = (idx >> 8) & 7;
  int kt   = idx >> 11;
  int kb = (lane >> 4) * 8;
  int n  = (lane & 15) + 16 * nt;
  int k0 = 32 * kt + kb + 2 * i + (i >= 4 ? 8 : 0);
  dst[idx] = pkbf(src[(size_t)k0 * DD + n], src[(size_t)(k0 + 1) * DD + n]);
}

// ---------------------------------------------------------------------------
// Conditioned-norm scale/shift vectors: condv[tk][0][j]=1+scale, [tk][1][j]=shift
// tk: 0..1 = embed(b), 2..13 = edge(s,b), 14..25 = node(s,b)
// ---------------------------------------------------------------------------
__global__ void cond_kernel(const float* __restrict__ tau,
    const float* cw1, const float* cb1, const float* cws, const float* cbs,
    const float* cwb, const float* cbb,
    const float* ew1, const float* eb1, const float* ews, const float* ebs,
    const float* ewb, const float* ebb,
    const float* nw1, const float* nb1, const float* nws, const float* nbs,
    const float* nwb, const float* nbb,
    float* __restrict__ condv) {
  int tk = blockIdx.x;
  int j  = threadIdx.x;
  const float *w1, *b1, *ws, *bs, *wb, *bb;
  int b;
  if (tk < 2) {
    b = tk; w1 = cw1; b1 = cb1; ws = cws; bs = cbs; wb = cwb; bb = cbb;
  } else if (tk < 14) {
    int s = (tk - 2) >> 1; b = (tk - 2) & 1;
    w1 = ew1 + s * 16; b1 = eb1 + s * 16; ws = ews + s * 2048; bs = ebs + s * 128;
    wb = ewb + s * 2048; bb = ebb + s * 128;
  } else {
    int s = (tk - 14) >> 1; b = (tk - 14) & 1;
    w1 = nw1 + s * 16; b1 = nb1 + s * 16; ws = nws + s * 2048; bs = nbs + s * 128;
    wb = nwb + s * 2048; bb = nbb + s * 128;
  }
  __shared__ float hh[16];
  if (j < 16) hh[j] = swishf(tau[b] * w1[j] + b1[j]);
  __syncthreads();
  float sc = bs[j], sh = bb[j];
#pragma unroll
  for (int t = 0; t < 16; ++t) { sc += hh[t] * ws[t * 128 + j]; sh += hh[t] * wb[t * 128 + j]; }
  condv[tk * 256 + j]       = 1.0f + sc;
  condv[tk * 256 + 128 + j] = sh;
}

__global__ void zero_kernel(float* __restrict__ p, size_t n4) {
  size_t i = (size_t)blockIdx.x * blockDim.x + threadIdx.x;
  size_t stride = (size_t)gridDim.x * blockDim.x;
  f32x4 z = {0.f, 0.f, 0.f, 0.f};
  for (size_t k = i; k < n4; k += stride) ((f32x4*)p)[k] = z;
}

__global__ void count_kernel(const int* __restrict__ receivers, float* __restrict__ counts) {
  int idx = blockIdx.x * 256 + threadIdx.x;
  if (idx >= NB * NE) return;
  int b = idx / NE;
  atomicAdd(&counts[(size_t)b * NN + receivers[idx]], 1.0f);
}

// ---------------------------------------------------------------------------
// Edge embedding: edges = cond_ln(swish(x@W1+b1)@W2+b2), x = [E,4] raw features
// One wave per 16-edge tile; GEMM1 (K=4) in VALU, GEMM2 (K=128) in WMMA bf16.
// ---------------------------------------------------------------------------
__global__ __launch_bounds__(256) void edge_embed_kernel(
    const float* __restrict__ ef, float* __restrict__ edges,
    const float* __restrict__ W1, const float* __restrict__ b1,
    const u32x4* __restrict__ w2p, const float* __restrict__ b2,
    const float* __restrict__ condv) {
  __shared__ __align__(16) u16 hbuf[8][16][DD];
  const int tid = threadIdx.x, wv = tid >> 5, lane = tid & 31;
  const int hi = lane >> 4, nl = lane & 15, kb = 8 * hi;
  int wid = blockIdx.x * 8 + wv;
  int b = wid / 10000;
  int e0 = (wid - b * 10000) * 16;

  float xf[8][4];
#pragma unroll
  for (int i = 0; i < 8; ++i) {
    size_t base = ((size_t)b * NE + e0 + i + 8 * hi) * 4;
#pragma unroll
    for (int k = 0; k < 4; ++k) xf[i][k] = ef[base + k];
  }
#pragma unroll
  for (int nt = 0; nt < 8; ++nt) {
    int col = nt * 16 + nl;
    float w0 = W1[col], w1v = W1[128 + col], w2v = W1[256 + col], w3 = W1[384 + col];
    float bb = b1[col];
#pragma unroll
    for (int i = 0; i < 8; ++i) {
      float v = bb + xf[i][0] * w0 + xf[i][1] * w1v + xf[i][2] * w2v + xf[i][3] * w3;
      hbuf[wv][i + 8 * hi][col] = __builtin_bit_cast(u16, (__bf16)swishf(v));
    }
  }

  f32x8 acc[8] = {};
#pragma unroll
  for (int t = 0; t < 4; ++t) {
    Frag16 af;
    __builtin_memcpy(&af.q[0], &hbuf[wv][nl][32 * t + kb], 16);
    __builtin_memcpy(&af.q[1], &hbuf[wv][nl][32 * t + kb + 16], 16);
    const u32x4* wt = w2p + (size_t)t * 512 + lane * 2;
#pragma unroll
    for (int nt = 0; nt < 8; ++nt) {
      Frag16 bf; bf.q[0] = wt[nt * 64]; bf.q[1] = wt[nt * 64 + 1];
      acc[nt] = wmma_bf16(af.v, bf.v, acc[nt]);
    }
  }

  float sum[8] = {}, ssq[8] = {}, mean[8], rstd[8];
#pragma unroll
  for (int nt = 0; nt < 8; ++nt) {
    float bb = b2[nt * 16 + nl];
#pragma unroll
    for (int i = 0; i < 8; ++i) {
      float v = acc[nt][i] + bb;
      acc[nt][i] = v; sum[i] += v; ssq[i] += v * v;
    }
  }
#pragma unroll
  for (int i = 0; i < 8; ++i) {
    float s1 = sum[i], s2 = ssq[i];
#pragma unroll
    for (int m = 1; m < 16; m <<= 1) { s1 += __shfl_xor(s1, m, 32); s2 += __shfl_xor(s2, m, 32); }
    float mu = s1 * (1.f / 128.f);
    mean[i] = mu;
    rstd[i] = __builtin_amdgcn_rsqf(s2 * (1.f / 128.f) - mu * mu + 1e-5f);
  }
  const float* cv = condv + b * 256;
#pragma unroll
  for (int nt = 0; nt < 8; ++nt) {
    int col = nt * 16 + nl;
    float sc = cv[col], sh = cv[128 + col];
#pragma unroll
    for (int i = 0; i < 8; ++i) {
      size_t off = ((size_t)b * NE + e0 + i + 8 * hi) * DD + col;
      edges[off] = (acc[nt][i] - mean[i]) * rstd[i] * sc + sh;
    }
  }
}

// ---------------------------------------------------------------------------
// Fused edge step: gather snd/rcv, GEMM1 [16x384]x[384x128] (WMMA bf16),
// swish, GEMM2 [16x128]x[128x128], cond-LN, edges += e_upd, atomic scatter.
// ---------------------------------------------------------------------------
__global__ __launch_bounds__(256) void edge_step_kernel(
    const float* __restrict__ nodes, float* __restrict__ edges,
    float* __restrict__ agg,
    const int* __restrict__ senders, const int* __restrict__ receivers,
    const u32x4* __restrict__ w1p, const u32x4* __restrict__ w2p,
    const float* __restrict__ be1, const float* __restrict__ be2,
    const float* __restrict__ condv, int s) {
  __shared__ __align__(16) u16 hbuf[8][16][DD];
  const int tid = threadIdx.x, wv = tid >> 5, lane = tid & 31;
  const int hi = lane >> 4, nl = lane & 15, kb = 8 * hi;
  int wid = blockIdx.x * 8 + wv;
  int b = wid / 10000;
  int e0 = (wid - b * 10000) * 16;

  // A-fragment row for this lane (m = nl)
  const float* erow = edges + ((size_t)b * NE + e0 + nl) * DD;
  int sidx = senders[(size_t)b * NE + e0 + nl];
  int ridx = receivers[(size_t)b * NE + e0 + nl];
  const float* srow = nodes + ((size_t)b * NN + sidx) * DD;
  const float* rrow = nodes + ((size_t)b * NN + ridx) * DD;

  f32x8 acc[8] = {};
  const u32x4* w1s = w1p + (size_t)s * 6144;  // 12 ktiles * 8 ntiles * 32 * 2
#pragma unroll
  for (int t = 0; t < 12; ++t) {
    const float* base = (t < 4) ? (erow + 32 * t)
                      : (t < 8) ? (srow + 32 * t - 128)
                                : (rrow + 32 * t - 256);
    bf16x16 a;
    setA8(a, 0, base + kb, 1.f);
    setA8(a, 8, base + kb + 16, 1.f);
    const u32x4* wt = w1s + (size_t)t * 512 + lane * 2;
#pragma unroll
    for (int nt = 0; nt < 8; ++nt) {
      Frag16 bf; bf.q[0] = wt[nt * 64]; bf.q[1] = wt[nt * 64 + 1];
      acc[nt] = wmma_bf16(a, bf.v, acc[nt]);
    }
  }

  const float* b1 = be1 + s * DD;
#pragma unroll
  for (int nt = 0; nt < 8; ++nt) {
    float bb = b1[nt * 16 + nl];
#pragma unroll
    for (int i = 0; i < 8; ++i)
      hbuf[wv][i + 8 * hi][nt * 16 + nl] =
          __builtin_bit_cast(u16, (__bf16)swishf(acc[nt][i] + bb));
  }

  f32x8 acc2[8] = {};
  const u32x4* w2s = w2p + (size_t)s * 2048;  // 4 ktiles * 8 ntiles * 32 * 2
#pragma unroll
  for (int t = 0; t < 4; ++t) {
    Frag16 af;
    __builtin_memcpy(&af.q[0], &hbuf[wv][nl][32 * t + kb], 16);
    __builtin_memcpy(&af.q[1], &hbuf[wv][nl][32 * t + kb + 16], 16);
    const u32x4* wt = w2s + (size_t)t * 512 + lane * 2;
#pragma unroll
    for (int nt = 0; nt < 8; ++nt) {
      Frag16 bf; bf.q[0] = wt[nt * 64]; bf.q[1] = wt[nt * 64 + 1];
      acc2[nt] = wmma_bf16(af.v, bf.v, acc2[nt]);
    }
  }

  const float* b2 = be2 + s * DD;
  float sum[8] = {}, ssq[8] = {}, mean[8], rstd[8];
#pragma unroll
  for (int nt = 0; nt < 8; ++nt) {
    float bb = b2[nt * 16 + nl];
#pragma unroll
    for (int i = 0; i < 8; ++i) {
      float v = acc2[nt][i] + bb;
      acc2[nt][i] = v; sum[i] += v; ssq[i] += v * v;
    }
  }
#pragma unroll
  for (int i = 0; i < 8; ++i) {
    float s1 = sum[i], s2 = ssq[i];
#pragma unroll
    for (int m = 1; m < 16; m <<= 1) { s1 += __shfl_xor(s1, m, 32); s2 += __shfl_xor(s2, m, 32); }
    float mu = s1 * (1.f / 128.f);
    mean[i] = mu;
    rstd[i] = __builtin_amdgcn_rsqf(s2 * (1.f / 128.f) - mu * mu + 1e-5f);
  }

  const float* cv = condv + (2 + s * 2 + b) * 256;
  int rid[8];
#pragma unroll
  for (int i = 0; i < 8; ++i) rid[i] = receivers[(size_t)b * NE + e0 + i + 8 * hi];
#pragma unroll
  for (int nt = 0; nt < 8; ++nt) {
    int col = nt * 16 + nl;
    float sc = cv[col], sh = cv[128 + col];
#pragma unroll
    for (int i = 0; i < 8; ++i) {
      float up = (acc2[nt][i] - mean[i]) * rstd[i] * sc + sh;
      size_t eoff = ((size_t)b * NE + e0 + i + 8 * hi) * DD + col;
      edges[eoff] = edges[eoff] + up;               // residual edges += e_upd
      atomicAdd(&agg[((size_t)b * NN + rid[i]) * DD + col], up);
    }
  }
}

// ---------------------------------------------------------------------------
// Fused node step: n_in=[nodes|agg/count], MLP via WMMA, cond-LN, nodes += upd
// ---------------------------------------------------------------------------
__global__ __launch_bounds__(256) void node_step_kernel(
    float* __restrict__ nodes, const float* __restrict__ agg,
    const float* __restrict__ counts,
    const u32x4* __restrict__ w1p, const u32x4* __restrict__ w2p,
    const float* __restrict__ bn1, const float* __restrict__ bn2,
    const float* __restrict__ condv, int s) {
  __shared__ __align__(16) u16 hbuf[8][16][DD];
  const int tid = threadIdx.x, wv = tid >> 5, lane = tid & 31;
  const int hi = lane >> 4, nl = lane & 15, kb = 8 * hi;
  int wid = blockIdx.x * 8 + wv;
  if (wid >= NB * 1250) return;
  int b = wid / 1250;
  int n0 = (wid - b * 1250) * 16;

  int node = n0 + nl;
  const float* nrow = nodes + ((size_t)b * NN + node) * DD;
  const float* grow = agg + ((size_t)b * NN + node) * DD;
  float invc = __builtin_amdgcn_rcpf(fmaxf(counts[(size_t)b * NN + node], 1.0f));

  f32x8 acc[8] = {};
  const u32x4* w1s = w1p + (size_t)s * 4096;  // 8 ktiles * 8 ntiles * 32 * 2
#pragma unroll
  for (int t = 0; t < 8; ++t) {
    bf16x16 a;
    if (t < 4) {
      const float* p = nrow + 32 * t;
      setA8(a, 0, p + kb, 1.f); setA8(a, 8, p + kb + 16, 1.f);
    } else {
      const float* p = grow + 32 * t - 128;
      setA8(a, 0, p + kb, invc); setA8(a, 8, p + kb + 16, invc);
    }
    const u32x4* wt = w1s + (size_t)t * 512 + lane * 2;
#pragma unroll
    for (int nt = 0; nt < 8; ++nt) {
      Frag16 bf; bf.q[0] = wt[nt * 64]; bf.q[1] = wt[nt * 64 + 1];
      acc[nt] = wmma_bf16(a, bf.v, acc[nt]);
    }
  }

  const float* b1 = bn1 + s * DD;
#pragma unroll
  for (int nt = 0; nt < 8; ++nt) {
    float bb = b1[nt * 16 + nl];
#pragma unroll
    for (int i = 0; i < 8; ++i)
      hbuf[wv][i + 8 * hi][nt * 16 + nl] =
          __builtin_bit_cast(u16, (__bf16)swishf(acc[nt][i] + bb));
  }

  f32x8 acc2[8] = {};
  const u32x4* w2s = w2p + (size_t)s * 2048;
#pragma unroll
  for (int t = 0; t < 4; ++t) {
    Frag16 af;
    __builtin_memcpy(&af.q[0], &hbuf[wv][nl][32 * t + kb], 16);
    __builtin_memcpy(&af.q[1], &hbuf[wv][nl][32 * t + kb + 16], 16);
    const u32x4* wt = w2s + (size_t)t * 512 + lane * 2;
#pragma unroll
    for (int nt = 0; nt < 8; ++nt) {
      Frag16 bf; bf.q[0] = wt[nt * 64]; bf.q[1] = wt[nt * 64 + 1];
      acc2[nt] = wmma_bf16(af.v, bf.v, acc2[nt]);
    }
  }

  const float* b2 = bn2 + s * DD;
  float sum[8] = {}, ssq[8] = {}, mean[8], rstd[8];
#pragma unroll
  for (int nt = 0; nt < 8; ++nt) {
    float bb = b2[nt * 16 + nl];
#pragma unroll
    for (int i = 0; i < 8; ++i) {
      float v = acc2[nt][i] + bb;
      acc2[nt][i] = v; sum[i] += v; ssq[i] += v * v;
    }
  }
#pragma unroll
  for (int i = 0; i < 8; ++i) {
    float s1 = sum[i], s2 = ssq[i];
#pragma unroll
    for (int m = 1; m < 16; m <<= 1) { s1 += __shfl_xor(s1, m, 32); s2 += __shfl_xor(s2, m, 32); }
    float mu = s1 * (1.f / 128.f);
    mean[i] = mu;
    rstd[i] = __builtin_amdgcn_rsqf(s2 * (1.f / 128.f) - mu * mu + 1e-5f);
  }

  const float* cv = condv + (14 + s * 2 + b) * 256;
#pragma unroll
  for (int nt = 0; nt < 8; ++nt) {
    int col = nt * 16 + nl;
    float sc = cv[col], sh = cv[128 + col];
#pragma unroll
    for (int i = 0; i < 8; ++i) {
      float up = (acc2[nt][i] - mean[i]) * rstd[i] * sc + sh;
      size_t off = ((size_t)b * NN + n0 + i + 8 * hi) * DD + col;
      nodes[off] = nodes[off] + up;                 // residual nodes += n_upd
    }
  }
}

extern "C" void kernel_launch(void* const* d_in, const int* in_sizes, int n_in,
                              void* d_out, int out_size, void* d_ws, size_t ws_size,
                              hipStream_t stream) {
  (void)in_sizes; (void)n_in; (void)out_size; (void)ws_size;
  const float* rnode    = (const float*)d_in[0];
  const float* ef       = (const float*)d_in[1];
  const int* senders    = (const int*)d_in[2];
  const int* receivers  = (const int*)d_in[3];
  const float* tau      = (const float*)d_in[4];
  const float* Wemb1    = (const float*)d_in[5];
  const float* bemb1    = (const float*)d_in[6];
  const float* Wemb2    = (const float*)d_in[7];
  const float* bemb2    = (const float*)d_in[8];
  float* out = (float*)d_out;

  // workspace layout (all 16B aligned)
  float* edges  = (float*)d_ws;                       // B*E*128 f32
  float* agg    = edges  + (size_t)NB * NE * DD;      // B*N*128 f32
  float* counts = agg    + (size_t)NB * NN * DD;      // B*N f32
  float* condv  = counts + (size_t)NB * NN;           // 26*256 f32
  u32* emb2p = (u32*)(condv + 26 * 256);              // packed bf16 frags
  u32* we1p = emb2p + 128 * 64;
  u32* we2p = we1p + 6 * 384 * 64;
  u32* wn1p = we2p + 6 * 128 * 64;
  u32* wn2p = wn1p + 6 * 256 * 64;

  // 1) pack weights (f32 -> bf16 WMMA B-fragment layout); stacked K over steps
  pack_w_kernel<<<(128 * 64 + 255) / 256, 256, 0, stream>>>(Wemb2, emb2p, 128);
  pack_w_kernel<<<(2304 * 64 + 255) / 256, 256, 0, stream>>>((const float*)d_in[15], we1p, 2304);
  pack_w_kernel<<<(768 * 64 + 255) / 256, 256, 0, stream>>>((const float*)d_in[17], we2p, 768);
  pack_w_kernel<<<(1536 * 64 + 255) / 256, 256, 0, stream>>>((const float*)d_in[25], wn1p, 1536);
  pack_w_kernel<<<(768 * 64 + 255) / 256, 256, 0, stream>>>((const float*)d_in[27], wn2p, 768);

  // 2) conditioned-norm scale/shift vectors for embed + all steps/batches
  cond_kernel<<<26, 128, 0, stream>>>(tau,
      (const float*)d_in[9],  (const float*)d_in[10], (const float*)d_in[11],
      (const float*)d_in[12], (const float*)d_in[13], (const float*)d_in[14],
      (const float*)d_in[19], (const float*)d_in[20], (const float*)d_in[21],
      (const float*)d_in[22], (const float*)d_in[23], (const float*)d_in[24],
      (const float*)d_in[29], (const float*)d_in[30], (const float*)d_in[31],
      (const float*)d_in[32], (const float*)d_in[33], (const float*)d_in[34],
      condv);

  // 3) receiver degree counts
  zero_kernel<<<64, 256, 0, stream>>>(counts, (size_t)NB * NN / 4);
  count_kernel<<<(NB * NE + 255) / 256, 256, 0, stream>>>(receivers, counts);

  // 4) nodes <- rnode_features
  hipMemcpyAsync(out, rnode, (size_t)NB * NN * DD * sizeof(float),
                 hipMemcpyDeviceToDevice, stream);

  // 5) edge embedding
  edge_embed_kernel<<<2500, 256, 0, stream>>>(ef, edges, Wemb1, bemb1,
                                              (const u32x4*)emb2p, bemb2, condv);

  // 6) message-passing steps
  for (int s = 0; s < NS; ++s) {
    zero_kernel<<<512, 256, 0, stream>>>(agg, (size_t)NB * NN * DD / 4);
    edge_step_kernel<<<2500, 256, 0, stream>>>(out, edges, agg, senders, receivers,
        (const u32x4*)we1p, (const u32x4*)we2p,
        (const float*)d_in[16], (const float*)d_in[18], condv, s);
    node_step_kernel<<<313, 256, 0, stream>>>(out, agg, counts,
        (const u32x4*)wn1p, (const u32x4*)wn2p,
        (const float*)d_in[26], (const float*)d_in[28], condv, s);
  }
}
